// SLSTMCell_13975823581654
// MI455X (gfx1250) — compile-verified, matrix operand order checked by reading, not run on hardware
//
#include <hip/hip_runtime.h>
#include <hip/hip_bf16.h>

namespace {

constexpr int Bc = 4096, INc = 1024, HSc = 256, HIDc = 2048, GATEc = 8192;
constexpr int BM = 128;   // rows per workgroup
constexpr int BN = 64;    // hidden (j) columns per workgroup (per gate)
constexpr int BK = 32;    // K tile (matches wmma 16x16x32)
constexpr int LDT = 40;   // padded LDS stride in bf16 elems (80B: conflict-free, 16B aligned)

typedef __attribute__((ext_vector_type(16))) __bf16          bf16x16;
typedef __attribute__((ext_vector_type(8)))  float           f32x8;
typedef __attribute__((ext_vector_type(4)))  float           f32x4;
typedef __attribute__((ext_vector_type(8)))  unsigned short  u16x8;
typedef int v4i __attribute__((vector_size(16)));   // matches async-lds builtin pointee type

// ---------------- common helpers ----------------

__device__ __forceinline__ unsigned short cvt_bf16(float f) {
  union { float f; unsigned u; } v; v.f = f;
  unsigned r = v.u + 0x7FFFu + ((v.u >> 16) & 1u);   // round to nearest even
  return (unsigned short)(r >> 16);
}

// Fragment (wave32, 16x16x32 bf16): lane L holds dim index (L&15),
// halves 0..7 = K kb..kb+7, halves 8..15 = K kb+16..kb+23, kb=(L>>4)*8.
__device__ __forceinline__ bf16x16 load_frag(const unsigned short* p) {
  union { bf16x16 v; u16x8 h[2]; } u;
  u.h[0] = *(const u16x8*)(p);
  u.h[1] = *(const u16x8*)(p + 16);
  return u.v;
}

__device__ __forceinline__ float fast_sigmoid(float x) {
  return 1.0f / (1.0f + __expf(-x));
}
__device__ __forceinline__ float fast_tanh(float x) {
  float t = __expf(-2.0f * fabsf(x));
  float r = (1.0f - t) / (1.0f + t);
  return copysignf(r, x);
}

// ---- async global->LDS copy (CDNA5 GLOBAL_LOAD_ASYNC_TO_LDS_B128, ASYNCcnt) ----
#if defined(__AMDGCN__) && __has_builtin(__builtin_amdgcn_global_load_async_to_lds_b128)
#define SLSTM_ASYNC 1
#else
#define SLSTM_ASYNC 0
#endif

__device__ __forceinline__ void cp16(const unsigned short* g, unsigned short* l) {
#if SLSTM_ASYNC
  __builtin_amdgcn_global_load_async_to_lds_b128(
      (__attribute__((address_space(1))) v4i*)(void*)g,
      (__attribute__((address_space(3))) v4i*)(void*)l, 0, 0);
#else
  *(u16x8*)l = *(const u16x8*)g;
#endif
}

__device__ __forceinline__ void staging_wait() {
#if SLSTM_ASYNC
#if __has_builtin(__builtin_amdgcn_s_wait_asynccnt)
  __builtin_amdgcn_s_wait_asynccnt(0);
#else
  asm volatile("s_wait_asynccnt 0x0" ::: "memory");
#endif
#endif
}

// Stage A tile (128 rows x 32 k bf16) from row-major global into LDS [row][k]
__device__ __forceinline__ void stage_tile_A(const unsigned short* __restrict__ g,
                                             size_t gstride, unsigned short* l, int tid) {
  #pragma unroll
  for (int it = 0; it < 2; ++it) {
    int idx = tid + it * 256;       // 512 chunks of 8 bf16
    int row = idx >> 2;             // 0..127
    int c16 = (idx & 3) << 3;       // 0,8,16,24
    cp16(g + (size_t)row * gstride + c16, l + row * LDT + c16);
  }
}
// Stage B tile (64 n-rows x 32 k bf16) from row-major (already transposed) global
__device__ __forceinline__ void stage_tile_B(const unsigned short* __restrict__ g,
                                             size_t gstride, unsigned short* l, int tid) {
  int row = tid >> 2;               // 0..63
  int c16 = (tid & 3) << 3;
  cp16(g + (size_t)row * gstride + c16, l + row * LDT + c16);
}

// ---------------- pre-pass kernels ----------------

// f32 -> bf16, 8 elems/thread
__global__ __launch_bounds__(256) void cvt_bf16_kernel(const float* __restrict__ in,
                                                       unsigned short* __restrict__ out, int n8) {
  int i = blockIdx.x * blockDim.x + threadIdx.x;
  if (i < n8) {
    f32x4 a = *(const f32x4*)(in + (size_t)i * 8);
    f32x4 b = *(const f32x4*)(in + (size_t)i * 8 + 4);
    u16x8 o;
    o[0] = cvt_bf16(a.x); o[1] = cvt_bf16(a.y); o[2] = cvt_bf16(a.z); o[3] = cvt_bf16(a.w);
    o[4] = cvt_bf16(b.x); o[5] = cvt_bf16(b.y); o[6] = cvt_bf16(b.z); o[7] = cvt_bf16(b.w);
    *(u16x8*)(out + (size_t)i * 8) = o;
  }
}

// batched 2D transpose with f32->bf16: out[b][c][r] = cvt(in[b][r][c]); rows,cols % 32 == 0
__global__ __launch_bounds__(256) void transpose_cvt_kernel(const float* __restrict__ in,
                                                            unsigned short* __restrict__ out,
                                                            int rows, int cols) {
  __shared__ unsigned short t[32][33];
  const int tx = threadIdx.x;                 // 0..31
  const int ty = threadIdx.y;                 // 0..7
  const int c0 = blockIdx.x * 32;
  const int r0 = blockIdx.y * 32;
  const size_t base = (size_t)blockIdx.z * rows * cols;
  #pragma unroll
  for (int i = 0; i < 4; ++i) {
    int r = r0 + ty + i * 8;
    t[ty + i * 8][tx] = cvt_bf16(in[base + (size_t)r * cols + c0 + tx]);
  }
  __syncthreads();
  #pragma unroll
  for (int i = 0; i < 4; ++i) {
    int c = c0 + ty + i * 8;
    out[base + (size_t)c * rows + r0 + tx] = t[tx][ty + i * 8];
  }
}

// ---------------- main fused sLSTM kernel (pre-converted bf16 operands) ----------------
// xb:  [B][IN] bf16            hb:  [B][HID] bf16
// WiT: [GATE][IN] bf16 (W_i^T) WhT: [NH][4HS][HS] bf16 (per-head W_h^T)
__global__ __launch_bounds__(256, 1) void slstm_wmma_bf16_kernel(
    const unsigned short* __restrict__ xb,  const unsigned short* __restrict__ hb,
    const unsigned short* __restrict__ WiT, const unsigned short* __restrict__ WhT,
    const float* __restrict__ cprev, const float* __restrict__ mprev,
    const float* __restrict__ nprev, const float* __restrict__ bi,
    const float* __restrict__ bh,    float* __restrict__ out) {
  __shared__ __align__(16) unsigned short Xs[2][BM * LDT];       // A tiles, double buffered
  __shared__ __align__(16) unsigned short Ws[2][4 * BN * LDT];   // B tiles (4 gate slots)

  const int tid  = threadIdx.x;
  const int lane = tid & 31;
  const int wave = tid >> 5;
  const int wm   = wave & 3;
  const int wn   = wave >> 2;
  const int b0   = blockIdx.y * BM;
  const int j0   = blockIdx.x * BN;
  const int h0   = j0 >> 10;
  const int jc   = j0 & 1023;

  const f32x8 zero = {0.f, 0.f, 0.f, 0.f, 0.f, 0.f, 0.f, 0.f};
  f32x8 acc[4][2][2];
  #pragma unroll
  for (int q = 0; q < 4; ++q)
    #pragma unroll
    for (int a = 0; a < 2; ++a)
      #pragma unroll
      for (int b2 = 0; b2 < 2; ++b2) acc[q][a][b2] = zero;

  const int r16 = lane & 15;
  const int kb  = (lane >> 4) << 3;

  const unsigned short* xrow = xb + (size_t)b0 * INc;

  // =========== Phase 1: x @ W_i, K = 1024, 4 gate column blocks, pipelined ===========
  stage_tile_A(xrow, INc, Xs[0], tid);
  #pragma unroll
  for (int q = 0; q < 4; ++q)
    stage_tile_B(WiT + (size_t)(q * HIDc + j0) * INc, INc, &Ws[0][q * BN * LDT], tid);
  staging_wait();
  __syncthreads();

  for (int s = 0; s < INc / BK; ++s) {
    const int cur = s & 1, nxt = cur ^ 1;
    if (s + 1 < INc / BK) {
      const int kk = (s + 1) * BK;
      stage_tile_A(xrow + kk, INc, Xs[nxt], tid);
      #pragma unroll
      for (int q = 0; q < 4; ++q)
        stage_tile_B(WiT + (size_t)(q * HIDc + j0) * INc + kk, INc, &Ws[nxt][q * BN * LDT], tid);
    }
    const unsigned short* Xc = Xs[cur];
    const unsigned short* Wc = Ws[cur];
    bf16x16 af0 = load_frag(&Xc[(wm * 32 +  0 + r16) * LDT + kb]);
    bf16x16 af1 = load_frag(&Xc[(wm * 32 + 16 + r16) * LDT + kb]);
    #pragma unroll
    for (int q = 0; q < 4; ++q) {
      #pragma unroll
      for (int b2 = 0; b2 < 2; ++b2) {
        bf16x16 bf = load_frag(&Wc[q * BN * LDT + (wn * 32 + b2 * 16 + r16) * LDT + kb]);
        acc[q][0][b2] = __builtin_amdgcn_wmma_f32_16x16x32_bf16(
            false, af0, false, bf, (short)0, acc[q][0][b2], false, false);
        acc[q][1][b2] = __builtin_amdgcn_wmma_f32_16x16x32_bf16(
            false, af1, false, bf, (short)0, acc[q][1][b2], false, false);
      }
    }
    staging_wait();
    __syncthreads();
  }

  // =========== Phase 2: per-gate h_head @ W_h[head], K = 256 each, pipelined ===========
  // step s in [0,32): gate q = s>>3, kk = (s&7)*BK, head = h0 + 2q
  stage_tile_A(hb + (size_t)b0 * HIDc + h0 * HSc, HIDc, Xs[0], tid);
  stage_tile_B(WhT + ((size_t)h0 * (4 * HSc) + jc) * HSc, HSc, Ws[0], tid);
  staging_wait();
  __syncthreads();

  #pragma unroll
  for (int q = 0; q < 4; ++q) {            // q static -> acc[q] stays in registers
    for (int kki = 0; kki < 8; ++kki) {
      const int s   = q * 8 + kki;
      const int cur = s & 1, nxt = cur ^ 1;
      if (s + 1 < 32) {
        const int sn   = s + 1;
        const int qn   = sn >> 3;
        const int kkn  = (sn & 7) * BK;
        const int hdn  = h0 + 2 * qn;
        stage_tile_A(hb + (size_t)b0 * HIDc + hdn * HSc + kkn, HIDc, Xs[nxt], tid);
        stage_tile_B(WhT + ((size_t)hdn * (4 * HSc) + jc) * HSc + kkn, HSc, Ws[nxt], tid);
      }
      const unsigned short* Xc = Xs[cur];
      const unsigned short* Wc = Ws[cur];
      bf16x16 af0 = load_frag(&Xc[(wm * 32 +  0 + r16) * LDT + kb]);
      bf16x16 af1 = load_frag(&Xc[(wm * 32 + 16 + r16) * LDT + kb]);
      #pragma unroll
      for (int b2 = 0; b2 < 2; ++b2) {
        bf16x16 bf = load_frag(&Wc[(wn * 32 + b2 * 16 + r16) * LDT + kb]);
        acc[q][0][b2] = __builtin_amdgcn_wmma_f32_16x16x32_bf16(
            false, af0, false, bf, (short)0, acc[q][0][b2], false, false);
        acc[q][1][b2] = __builtin_amdgcn_wmma_f32_16x16x32_bf16(
            false, af1, false, bf, (short)0, acc[q][1][b2], false, false);
      }
      staging_wait();
      __syncthreads();
    }
  }

  // =========== Epilogue: fused sLSTM gate math, 4 outputs ===========
  const size_t N = (size_t)Bc * HIDc;
  float* outH = out;
  float* outC = out + N;
  float* outM = out + 2 * N;
  float* outN = out + 3 * N;

  #pragma unroll
  for (int b2 = 0; b2 < 2; ++b2) {
    const int j = j0 + wn * 32 + b2 * 16 + r16;
    const float bias_i = bi[0 * HIDc + j] + bh[0 * HIDc + j];
    const float bias_f = bi[1 * HIDc + j] + bh[1 * HIDc + j];
    const float bias_z = bi[2 * HIDc + j] + bh[2 * HIDc + j];
    const float bias_o = bi[3 * HIDc + j] + bh[3 * HIDc + j];
    #pragma unroll
    for (int a = 0; a < 2; ++a) {
      #pragma unroll
      for (int r = 0; r < 8; ++r) {
        const int row = b0 + wm * 32 + a * 16 + (lane >> 4) * 8 + r;
        const size_t idx = (size_t)row * HIDc + j;
        const float it_ = acc[0][a][b2][r] + bias_i;
        const float ft_ = acc[1][a][b2][r] + bias_f;
        const float zt_ = acc[2][a][b2][r] + bias_z;
        const float ot_ = acc[3][a][b2][r] + bias_o;
        const float mv = mprev[idx], cv = cprev[idx], nv = nprev[idx];
        const float mn = fmaxf(ft_ + mv, it_);
        const float iv = __expf(it_ - mn);
        const float fv = __expf(ft_ - mn + mv);
        const float zv = fast_tanh(zt_);
        const float ov = fast_sigmoid(ot_);
        const float cn = fv * cv + iv * zv;
        const float nn = fv * nv + iv;
        outH[idx] = ov * (cn / nn);
        outC[idx] = cn;
        outM[idx] = mn;
        outN[idx] = nn;
      }
    }
  }
}

// ---------------- fallback: fully fused fp32-input kernel (round-1 style) ----------------
__global__ __launch_bounds__(256, 1) void slstm_wmma_fused_kernel(
    const float* __restrict__ x,     const float* __restrict__ hprev,
    const float* __restrict__ cprev, const float* __restrict__ mprev,
    const float* __restrict__ nprev, const float* __restrict__ Wi,
    const float* __restrict__ bi,    const float* __restrict__ Wh,
    const float* __restrict__ bh,    float* __restrict__ out) {
  __shared__ __align__(16) unsigned short Xs[BM * LDT];
  __shared__ __align__(16) unsigned short Ws[4][BN * LDT];

  const int tid  = threadIdx.x;
  const int lane = tid & 31;
  const int wave = tid >> 5;
  const int wm   = wave & 3;
  const int wn   = wave >> 2;
  const int b0   = blockIdx.y * BM;
  const int j0   = blockIdx.x * BN;
  const int h0   = j0 >> 10;
  const int jc   = j0 & 1023;

  const f32x8 zero = {0.f, 0.f, 0.f, 0.f, 0.f, 0.f, 0.f, 0.f};
  f32x8 acc[4][2][2];
  #pragma unroll
  for (int q = 0; q < 4; ++q)
    #pragma unroll
    for (int a = 0; a < 2; ++a)
      #pragma unroll
      for (int b2 = 0; b2 < 2; ++b2) acc[q][a][b2] = zero;

  const int r16 = lane & 15;
  const int kb  = (lane >> 4) << 3;

  for (int kk = 0; kk < INc; kk += BK) {
    #pragma unroll
    for (int it = 0; it < 4; ++it) {
      int idx = tid + it * 256, row = idx >> 3, col4 = (idx & 7) << 2;
      f32x4 v = *(const f32x4*)(x + (size_t)(b0 + row) * INc + kk + col4);
      unsigned short* d = &Xs[row * LDT + col4];
      d[0] = cvt_bf16(v.x); d[1] = cvt_bf16(v.y); d[2] = cvt_bf16(v.z); d[3] = cvt_bf16(v.w);
    }
    #pragma unroll
    for (int q = 0; q < 4; ++q)
      #pragma unroll
      for (int it = 0; it < 2; ++it) {
        int idx = tid + it * 256, krow = idx >> 4, col4 = (idx & 15) << 2;
        f32x4 v = *(const f32x4*)(Wi + (size_t)(kk + krow) * GATEc + q * HIDc + j0 + col4);
        Ws[q][(col4 + 0) * LDT + krow] = cvt_bf16(v.x);
        Ws[q][(col4 + 1) * LDT + krow] = cvt_bf16(v.y);
        Ws[q][(col4 + 2) * LDT + krow] = cvt_bf16(v.z);
        Ws[q][(col4 + 3) * LDT + krow] = cvt_bf16(v.w);
      }
    __syncthreads();
    bf16x16 af0 = load_frag(&Xs[(wm * 32 +  0 + r16) * LDT + kb]);
    bf16x16 af1 = load_frag(&Xs[(wm * 32 + 16 + r16) * LDT + kb]);
    #pragma unroll
    for (int q = 0; q < 4; ++q)
      #pragma unroll
      for (int b2 = 0; b2 < 2; ++b2) {
        bf16x16 bf = load_frag(&Ws[q][(wn * 32 + b2 * 16 + r16) * LDT + kb]);
        acc[q][0][b2] = __builtin_amdgcn_wmma_f32_16x16x32_bf16(false, af0, false, bf, (short)0, acc[q][0][b2], false, false);
        acc[q][1][b2] = __builtin_amdgcn_wmma_f32_16x16x32_bf16(false, af1, false, bf, (short)0, acc[q][1][b2], false, false);
      }
    __syncthreads();
  }

  for (int q = 0; q < 4; ++q) {
    const int head = h0 + 2 * q;
    for (int kk = 0; kk < HSc; kk += BK) {
      #pragma unroll
      for (int it = 0; it < 4; ++it) {
        int idx = tid + it * 256, row = idx >> 3, col4 = (idx & 7) << 2;
        f32x4 v = *(const f32x4*)(hprev + (size_t)(b0 + row) * HIDc + head * HSc + kk + col4);
        unsigned short* d = &Xs[row * LDT + col4];
        d[0] = cvt_bf16(v.x); d[1] = cvt_bf16(v.y); d[2] = cvt_bf16(v.z); d[3] = cvt_bf16(v.w);
      }
      #pragma unroll
      for (int it = 0; it < 2; ++it) {
        int idx = tid + it * 256, krow = idx >> 4, col4 = (idx & 15) << 2;
        f32x4 v = *(const f32x4*)(Wh + ((size_t)head * HSc + kk + krow) * (4 * HSc) + jc + col4);
        Ws[0][(col4 + 0) * LDT + krow] = cvt_bf16(v.x);
        Ws[0][(col4 + 1) * LDT + krow] = cvt_bf16(v.y);
        Ws[0][(col4 + 2) * LDT + krow] = cvt_bf16(v.z);
        Ws[0][(col4 + 3) * LDT + krow] = cvt_bf16(v.w);
      }
      __syncthreads();
      bf16x16 af0 = load_frag(&Xs[(wm * 32 +  0 + r16) * LDT + kb]);
      bf16x16 af1 = load_frag(&Xs[(wm * 32 + 16 + r16) * LDT + kb]);
      #pragma unroll
      for (int b2 = 0; b2 < 2; ++b2) {
        bf16x16 bf = load_frag(&Ws[0][(wn * 32 + b2 * 16 + r16) * LDT + kb]);
        acc[q][0][b2] = __builtin_amdgcn_wmma_f32_16x16x32_bf16(false, af0, false, bf, (short)0, acc[q][0][b2], false, false);
        acc[q][1][b2] = __builtin_amdgcn_wmma_f32_16x16x32_bf16(false, af1, false, bf, (short)0, acc[q][1][b2], false, false);
      }
      __syncthreads();
    }
  }

  const size_t N = (size_t)Bc * HIDc;
  float* outH = out;
  float* outC = out + N;
  float* outM = out + 2 * N;
  float* outN = out + 3 * N;
  #pragma unroll
  for (int b2 = 0; b2 < 2; ++b2) {
    const int j = j0 + wn * 32 + b2 * 16 + r16;
    const float bias_i = bi[0 * HIDc + j] + bh[0 * HIDc + j];
    const float bias_f = bi[1 * HIDc + j] + bh[1 * HIDc + j];
    const float bias_z = bi[2 * HIDc + j] + bh[2 * HIDc + j];
    const float bias_o = bi[3 * HIDc + j] + bh[3 * HIDc + j];
    #pragma unroll
    for (int a = 0; a < 2; ++a)
      #pragma unroll
      for (int r = 0; r < 8; ++r) {
        const int row = b0 + wm * 32 + a * 16 + (lane >> 4) * 8 + r;
        const size_t idx = (size_t)row * HIDc + j;
        const float it_ = acc[0][a][b2][r] + bias_i;
        const float ft_ = acc[1][a][b2][r] + bias_f;
        const float zt_ = acc[2][a][b2][r] + bias_z;
        const float ot_ = acc[3][a][b2][r] + bias_o;
        const float mv = mprev[idx], cv = cprev[idx], nv = nprev[idx];
        const float mn = fmaxf(ft_ + mv, it_);
        const float iv = __expf(it_ - mn);
        const float fv = __expf(ft_ - mn + mv);
        const float zv = fast_tanh(zt_);
        const float ov = fast_sigmoid(ot_);
        const float cn = fv * cv + iv * zv;
        const float nn = fv * nv + iv;
        outH[idx] = ov * (cn / nn);
        outC[idx] = cn;
        outM[idx] = mn;
        outN[idx] = nn;
      }
  }
}

// workspace layout (bytes)
constexpr size_t XB_OFF  = 0;                       // [B][IN]   bf16  8,388,608
constexpr size_t HB_OFF  = 8388608;                 // [B][HID]  bf16 16,777,216
constexpr size_t WIT_OFF = 25165824;                // [GATE][IN]  bf16 16,777,216
constexpr size_t WHT_OFF = 41943040;                // [NH][4HS][HS] bf16 4,194,304
constexpr size_t WS_NEEDED = 46137344;

} // namespace

extern "C" void kernel_launch(void* const* d_in, const int* in_sizes, int n_in,
                              void* d_out, int out_size, void* d_ws, size_t ws_size,
                              hipStream_t stream) {
  (void)in_sizes; (void)n_in; (void)out_size;
  const float* x  = (const float*)d_in[0];
  const float* h  = (const float*)d_in[1];
  const float* c  = (const float*)d_in[2];
  const float* m  = (const float*)d_in[3];
  const float* n  = (const float*)d_in[4];
  const float* Wi = (const float*)d_in[5];
  const float* bi = (const float*)d_in[6];
  const float* Wh = (const float*)d_in[7];
  const float* bh = (const float*)d_in[8];
  float* out = (float*)d_out;

  if (ws_size >= WS_NEEDED && d_ws != nullptr) {
    unsigned short* xb  = (unsigned short*)((char*)d_ws + XB_OFF);
    unsigned short* hb  = (unsigned short*)((char*)d_ws + HB_OFF);
    unsigned short* WiT = (unsigned short*)((char*)d_ws + WIT_OFF);
    unsigned short* WhT = (unsigned short*)((char*)d_ws + WHT_OFF);

    // pre-pass: convert / transpose once (memory-bound, ~138 MB total traffic)
    cvt_bf16_kernel<<<dim3((Bc * INc / 8) / 256), dim3(256), 0, stream>>>(x, xb, Bc * INc / 8);
    cvt_bf16_kernel<<<dim3((Bc * HIDc / 8) / 256), dim3(256), 0, stream>>>(h, hb, Bc * HIDc / 8);
    transpose_cvt_kernel<<<dim3(GATEc / 32, INc / 32, 1), dim3(32, 8), 0, stream>>>(Wi, WiT, INc, GATEc);
    transpose_cvt_kernel<<<dim3((4 * HSc) / 32, HSc / 32, 8), dim3(32, 8), 0, stream>>>(Wh, WhT, HSc, 4 * HSc);

    dim3 grid(HIDc / BN, Bc / BM);   // 32 x 32 workgroups
    slstm_wmma_bf16_kernel<<<grid, dim3(256), 0, stream>>>(
        xb, hb, WiT, WhT, c, m, n, bi, bh, out);
  } else {
    dim3 grid(HIDc / BN, Bc / BM);
    slstm_wmma_fused_kernel<<<grid, dim3(256), 0, stream>>>(
        x, h, c, m, n, Wi, bi, Wh, bh, out);
  }
}
